// cross_scale_trans_13168369729996
// MI455X (gfx1250) — compile-verified, matrix operand order checked by reading, not run on hardware
//
#include <hip/hip_runtime.h>

#define N_PTS   8192
#define C_IN    32
#define DMODEL  128
#define NHEAD   8
#define MNEIGH  16
#define DFFN    256
#define DHEAD   16

typedef _Float16 h16;
typedef __attribute__((ext_vector_type(16))) _Float16 v16h;
typedef __attribute__((ext_vector_type(8)))  _Float16 v8h;
typedef __attribute__((ext_vector_type(8)))  float    v8f;

__device__ __forceinline__ unsigned ballot32(bool p) {
#if __has_builtin(__builtin_amdgcn_ballot_w32)
  return __builtin_amdgcn_ballot_w32(p);
#else
  return (unsigned)__ballot(p);
#endif
}

// Load one lane's A/B fragment for v_wmma_f32_16x16x32_f16.
// Per ISA: lanes 0-15 hold K = kh0+{0..7} and kh0+16+{0..7} with kh0=0,
// lanes 16-31 with kh0=8.  p must already point at [row, k0 + kh0].
__device__ __forceinline__ v16h load_frag(const h16* p) {
  v8h lo = *(const v8h*)(p);
  v8h hi = *(const v8h*)(p + 16);
  v16h a;
#pragma unroll
  for (int i = 0; i < 8; ++i) { a[i] = lo[i]; a[i + 8] = hi[i]; }
  return a;
}

// One 16x16 f32 tile of A(16xK) * W(16xK)^T.  a_row / w_row are pre-offset by +kh0.
__device__ __forceinline__ v8f tile_gemm(const h16* a_row, bool a_zero,
                                         const h16* w_row, int K) {
  v8f acc = {};
  for (int k0 = 0; k0 < K; k0 += 32) {
    v16h a = {};
    if (!a_zero) a = load_frag(a_row + k0);
    v16h b = load_frag(w_row + k0);
    acc = __builtin_amdgcn_wmma_f32_16x16x32_f16(false, a, false, b, (short)0,
                                                 acc, false, false);
  }
  return acc;
}

// ---------------------------------------------------------------- f32 -> f16
__global__ void k_f2h(const float* __restrict__ s, h16* __restrict__ d, int n) {
  int i = blockIdx.x * blockDim.x + threadIdx.x;
  if (i < n) d[i] = (h16)s[i];
}

// ------------------------------------------------- embed: src = proj + posMLP
__global__ void __launch_bounds__(128)
k_embed(const float* __restrict__ feat, const int* __restrict__ coords,
        const float* __restrict__ Wp, const float* __restrict__ bp,
        const float* __restrict__ Wpe1, const float* __restrict__ bpe1,
        const float* __restrict__ Wpe2, const float* __restrict__ bpe2,
        h16* __restrict__ srcH, int* __restrict__ cpack) {
  __shared__ float f[C_IN];
  __shared__ float hmid[64];
  __shared__ float cn[3];
  const int n = blockIdx.x, d = threadIdx.x;
  if (d < C_IN) f[d] = feat[(long)n * C_IN + d];
  if (d < 3) cn[d] = (float)coords[n * 3 + d] * (1.f / 63.f);
  if (d == 0)
    cpack[n] = coords[n * 3] | (coords[n * 3 + 1] << 8) | (coords[n * 3 + 2] << 16);
  __syncthreads();
  if (d < 64) {
    float s = bpe1[d];
    for (int a = 0; a < 3; ++a) s += cn[a] * Wpe1[d * 3 + a];
    hmid[d] = fmaxf(s, 0.f);
  }
  __syncthreads();
  float s = bp[d] + bpe2[d];
  for (int c = 0; c < C_IN; ++c) s += f[c] * Wp[d * C_IN + c];
  for (int j = 0; j < 64; ++j) s += hmid[j] * Wpe2[d * 64 + j];
  srcH[(long)n * DMODEL + d] = (h16)s;
}

// -------------------------------- neighbor top-16 by (L1 dist asc, index asc)
__global__ void __launch_bounds__(256)
k_neigh(const int* __restrict__ cpack, int* __restrict__ nidx, int n) {
  __shared__ int sc[N_PTS];
  __shared__ int lists[8][3][MNEIGH];
  for (int i = threadIdx.x; i < n; i += blockDim.x) sc[i] = cpack[i];
  __syncthreads();
  const int lane = threadIdx.x & 31;
  const int wave = threadIdx.x >> 5;
  const int pt = blockIdx.x * 8 + wave;
  const int me = sc[pt];
  const int mx = me & 255, my = (me >> 8) & 255, mz = (me >> 16) & 255;
  int c0 = 0, c1 = 0, c2 = 0;
  const unsigned lower = (1u << lane) - 1u;
  for (int base = 0; base < n; base += 32) {
    int cc = sc[base + lane];
    int d = abs((cc & 255) - mx) + abs(((cc >> 8) & 255) - my) +
            abs(((cc >> 16) & 255) - mz);
    unsigned m0 = ballot32(d == 0);
    if (d == 0) { int pos = c0 + __popc(m0 & lower); if (pos < MNEIGH) lists[wave][0][pos] = base + lane; }
    c0 += __popc(m0);
    unsigned m1 = ballot32(d == 1);
    if (d == 1) { int pos = c1 + __popc(m1 & lower); if (pos < MNEIGH) lists[wave][1][pos] = base + lane; }
    c1 += __popc(m1);
    unsigned m2 = ballot32(d == 2);
    if (d == 2) { int pos = c2 + __popc(m2 & lower); if (pos < MNEIGH) lists[wave][2][pos] = base + lane; }
    c2 += __popc(m2);
  }
  c0 = min(c0, MNEIGH); c1 = min(c1, MNEIGH); c2 = min(c2, MNEIGH);
  if (lane < MNEIGH) {
    int v;
    if (lane < c0)                 v = lists[wave][0][lane];
    else if (lane < c0 + c1)       v = lists[wave][1][lane - c0];
    else if (lane < c0 + c1 + c2)  v = lists[wave][2][lane - c0 - c1];
    else                           v = -1;
    nidx[(long)pt * MNEIGH + lane] = v;
  }
}

// ------------- fused transformer block: 16 points (256 gathered rows) / block
__global__ void __launch_bounds__(256)
k_attn_block(const float* __restrict__ features,
             const h16* __restrict__ srcH, const int* __restrict__ nidx,
             const h16* Wq, const float* bq, const h16* Wk, const float* bk,
             const h16* Wv, const float* bv, const h16* Wo, const float* bo,
             const h16* W1, const float* b1, const h16* W2, const float* b2,
             const float* lng, const float* lnb,
             const h16* Wf1, const float* bf1, const h16* Wf2, const float* bf2,
             float* __restrict__ fused) {
  extern __shared__ char smem[];
  h16*  sWk  = (h16*)smem;            // 128 x 128 f16  (32 KB, staged weights)
  h16*  sWv  = sWk + 128 * DMODEL;    // 128 x 128 f16  (32 KB)
  h16*  sK   = sWv + 128 * DMODEL;    // 256 x 128 f16  (64 KB)
  h16*  sV   = sK  + 256 * DMODEL;    // 256 x 128 f16  (64 KB)
  h16*  sQ   = sV  + 256 * DMODEL;    // 16 x 128
  h16*  sAt  = sQ  + 16 * DMODEL;     // 16 x 128
  h16*  sO   = sAt + 16 * DMODEL;     // 16 x 128
  h16*  sT1  = sO  + 16 * DMODEL;     // 16 x 256
  h16*  sTg  = sT1 + 16 * DFFN;       // 16 x 128
  h16*  sCat = sTg + 16 * DMODEL;     // 16 x 64
  float* sO32 = (float*)(sCat + 16 * 64); // 16 x 128 f32

  const int lane = threadIdx.x & 31;
  const int wave = threadIdx.x >> 5;
  const int lr = lane & 15;
  const int lh = lane >> 4;
  const int kh0 = lh * 8;

  // Prefetch the L2-resident FFN weights this block will stream later.
  __builtin_prefetch((const char*)W1 + threadIdx.x * 128, 0, 0);
  __builtin_prefetch((const char*)W1 + (threadIdx.x + 256) * 128, 0, 0);
  __builtin_prefetch((const char*)W2 + threadIdx.x * 128, 0, 0);
  __builtin_prefetch((const char*)W2 + (threadIdx.x + 256) * 128, 0, 0);

  // Stage Wk / Wv into LDS (reused by 16 m-tiles each).
  {
    const int4* gk = (const int4*)Wk;
    const int4* gv = (const int4*)Wv;
    int4* lk = (int4*)sWk;
    int4* lv = (int4*)sWv;
    for (int i = threadIdx.x; i < (128 * DMODEL) / 8; i += 256) {
      lk[i] = gk[i];
      lv[i] = gv[i];
    }
  }
  __syncthreads();

  // Phase 1a: K and V tiles together — one gathered A fragment, two WMMAs.
  // Invalid neighbors (idx<0) give zero rows, bias still added (matches ref).
  for (int t = wave; t < 128; t += 8) {
    int mt = t >> 3, nt = t & 7;
    int idx = nidx[(long)blockIdx.x * 256 + mt * 16 + lr];
    const h16* a_row = srcH + (long)idx * DMODEL + kh0;
    const bool a_zero = idx < 0;
    const h16* wk_row = sWk + (nt * 16 + lr) * DMODEL + kh0;
    const h16* wv_row = sWv + (nt * 16 + lr) * DMODEL + kh0;
    v8f accK = {}, accV = {};
    for (int k0 = 0; k0 < DMODEL; k0 += 32) {
      v16h a = {};
      if (!a_zero) a = load_frag(a_row + k0);
      v16h bK = load_frag(wk_row + k0);
      v16h bV = load_frag(wv_row + k0);
      accK = __builtin_amdgcn_wmma_f32_16x16x32_f16(false, a, false, bK, (short)0,
                                                    accK, false, false);
      accV = __builtin_amdgcn_wmma_f32_16x16x32_f16(false, a, false, bV, (short)0,
                                                    accV, false, false);
    }
    float bbk = bk[nt * 16 + lr];
    float bbv = bv[nt * 16 + lr];
#pragma unroll
    for (int v = 0; v < 8; ++v) {
      int row = mt * 16 + v + 8 * lh;
      sK[row * DMODEL + nt * 16 + lr] = (h16)(accK[v] + bbk);
      sV[row * DMODEL + nt * 16 + lr] = (h16)(accV[v] + bbv);
    }
  }

  // Phase 1b: Q tile (one per wave; rows = m=0 neighbors, no reuse -> global W).
  {
    int nt = wave;
    int idx = nidx[((long)blockIdx.x * 16 + lr) * MNEIGH];
    v8f acc = tile_gemm(srcH + (long)idx * DMODEL + kh0, idx < 0,
                        Wq + (long)(nt * 16 + lr) * DMODEL + kh0, DMODEL);
    float bb = bq[nt * 16 + lr];
#pragma unroll
    for (int v = 0; v < 8; ++v)
      sQ[(v + 8 * lh) * DMODEL + nt * 16 + lr] = (h16)(acc[v] + bb);
  }
  __syncthreads();

  // Phase 2: softmax attention (2 points per wave; lanes mirror in 16-groups).
  for (int it = 0; it < 2; ++it) {
    int pl = wave * 2 + it;
    const v8h* k8 = (const v8h*)(sK + (pl * MNEIGH + lr) * DMODEL);
    const v8h* v8p = (const v8h*)(sV + (pl * MNEIGH + lr) * DMODEL);
    const v8h* q8 = (const v8h*)(sQ + pl * DMODEL);
#pragma unroll
    for (int h = 0; h < NHEAD; ++h) {
      v8h ka = k8[2 * h], kb = k8[2 * h + 1];
      v8h qa = q8[2 * h], qb = q8[2 * h + 1];
      float s = 0.f;
#pragma unroll
      for (int i = 0; i < 8; ++i)
        s += (float)ka[i] * (float)qa[i] + (float)kb[i] * (float)qb[i];
      s *= 0.25f;                             // 1/sqrt(DH)
      float mxv = s;
      mxv = fmaxf(mxv, __shfl_xor(mxv, 1, 32));
      mxv = fmaxf(mxv, __shfl_xor(mxv, 2, 32));
      mxv = fmaxf(mxv, __shfl_xor(mxv, 4, 32));
      mxv = fmaxf(mxv, __shfl_xor(mxv, 8, 32));
      float e = __expf(s - mxv);
      float den = e;
      den += __shfl_xor(den, 1, 32);
      den += __shfl_xor(den, 2, 32);
      den += __shfl_xor(den, 4, 32);
      den += __shfl_xor(den, 8, 32);
      float w = e / den;
      v8h va = v8p[2 * h], vb = v8p[2 * h + 1];
      float ov[DHEAD];
#pragma unroll
      for (int i = 0; i < 8; ++i) {
        ov[i] = w * (float)va[i];
        ov[i + 8] = w * (float)vb[i];
      }
#pragma unroll
      for (int m = 1; m <= 8; m <<= 1) {
#pragma unroll
        for (int i = 0; i < DHEAD; ++i) ov[i] += __shfl_xor(ov[i], m, 32);
      }
      if (lane == 0) {
        v8h oa, ob;
#pragma unroll
        for (int i = 0; i < 8; ++i) { oa[i] = (h16)ov[i]; ob[i] = (h16)ov[i + 8]; }
        *(v8h*)(sAt + pl * DMODEL + h * DHEAD) = oa;
        *(v8h*)(sAt + pl * DMODEL + h * DHEAD + 8) = ob;
      }
    }
  }
  __syncthreads();

  // Phase 3: O = At @ Wo^T + bo   (one 16x16 tile per wave)
  {
    int nt = wave;
    v8f acc = tile_gemm(sAt + lr * DMODEL + kh0, false,
                        Wo + (long)(nt * 16 + lr) * DMODEL + kh0, DMODEL);
#pragma unroll
    for (int v = 0; v < 8; ++v) {
      int row = v + 8 * lh, col = nt * 16 + lr;
      float val = acc[v] + bo[col];
      sO[row * DMODEL + col] = (h16)val;
      sO32[row * DMODEL + col] = val;
    }
  }
  __syncthreads();

  // Phase 4: T1 = relu(O @ Wffn1^T + b1)
  for (int t = wave; t < 16; t += 8) {
    v8f acc = tile_gemm(sO + lr * DMODEL + kh0, false,
                        W1 + (long)(t * 16 + lr) * DMODEL + kh0, DMODEL);
#pragma unroll
    for (int v = 0; v < 8; ++v) {
      int row = v + 8 * lh, col = t * 16 + lr;
      sT1[row * DFFN + col] = (h16)fmaxf(acc[v] + b1[col], 0.f);
    }
  }
  __syncthreads();

  // Phase 5: t = O + T1 @ Wffn2^T + b2   (accumulate into sO32)
  {
    int nt = wave;
    v8f acc = tile_gemm(sT1 + lr * DFFN + kh0, false,
                        W2 + (long)(nt * 16 + lr) * DFFN + kh0, DFFN);
#pragma unroll
    for (int v = 0; v < 8; ++v) {
      int row = v + 8 * lh, col = nt * 16 + lr;
      sO32[row * DMODEL + col] += acc[v] + b2[col];
    }
  }
  __syncthreads();

  // Phase 6: LayerNorm rows -> sTg (f16)
  for (int it = 0; it < 2; ++it) {
    int rl = wave * 2 + it;
    float t0[4];
#pragma unroll
    for (int i = 0; i < 4; ++i) t0[i] = sO32[rl * DMODEL + lane * 4 + i];
    float s = t0[0] + t0[1] + t0[2] + t0[3];
    for (int m = 16; m >= 1; m >>= 1) s += __shfl_xor(s, m, 32);
    float mean = s * (1.f / DMODEL);
    float q = 0.f;
#pragma unroll
    for (int i = 0; i < 4; ++i) { float dd = t0[i] - mean; q += dd * dd; }
    for (int m = 16; m >= 1; m >>= 1) q += __shfl_xor(q, m, 32);
    float rstd = rsqrtf(q * (1.f / DMODEL) + 1e-5f);
#pragma unroll
    for (int i = 0; i < 4; ++i) {
      int c = lane * 4 + i;
      sTg[rl * DMODEL + c] = (h16)((t0[i] - mean) * rstd * lng[c] + lnb[c]);
    }
  }
  __syncthreads();

  // Phase 7: sCat[:, :32] = features ; sCat[:, 32:] = Tg @ Wf1^T + bf1
  for (int i = threadIdx.x; i < 16 * C_IN; i += 256) {
    int rl = i >> 5, c = i & 31;
    sCat[rl * 64 + c] = (h16)features[((long)blockIdx.x * 16 + rl) * C_IN + c];
  }
  if (wave < 2) {
    int nt = wave;
    v8f acc = tile_gemm(sTg + lr * DMODEL + kh0, false,
                        Wf1 + (long)(nt * 16 + lr) * DMODEL + kh0, DMODEL);
#pragma unroll
    for (int v = 0; v < 8; ++v) {
      int row = v + 8 * lh, col = nt * 16 + lr;
      sCat[row * 64 + 32 + col] = (h16)(acc[v] + bf1[col]);
    }
  }
  __syncthreads();

  // Phase 8: fused_pre = Cat @ Wf2^T + bf2 -> global (pre-BatchNorm)
  if (wave < 2) {
    int nt = wave;
    v8f acc = tile_gemm(sCat + lr * 64 + kh0, false,
                        Wf2 + (long)(nt * 16 + lr) * 64 + kh0, 64);
#pragma unroll
    for (int v = 0; v < 8; ++v) {
      int row = v + 8 * lh, col = nt * 16 + lr;
      fused[((long)blockIdx.x * 16 + row) * C_IN + col] = acc[v] + bf2[col];
    }
  }
}

// ------------------------------------------------ BatchNorm stats (per channel)
__global__ void __launch_bounds__(256)
k_bnstats(const float* __restrict__ fp, float* __restrict__ stats, int n) {
  __shared__ float as[256], as2[256];
  int c = blockIdx.x, tid = threadIdx.x;
  float s = 0.f, s2 = 0.f;
  for (int r = tid; r < n; r += 256) {
    float v = fp[(long)r * C_IN + c];
    s += v; s2 += v * v;
  }
  as[tid] = s; as2[tid] = s2;
  __syncthreads();
  for (int step = 128; step > 0; step >>= 1) {
    if (tid < step) { as[tid] += as[tid + step]; as2[tid] += as2[tid + step]; }
    __syncthreads();
  }
  if (tid == 0) {
    float mean = as[0] / n;
    float var = as2[0] / n - mean * mean;
    stats[c] = mean;
    stats[C_IN + c] = rsqrtf(var + 1e-5f);
  }
}

__global__ void k_bnapply(const float* __restrict__ fp, const float* __restrict__ stats,
                          const float* __restrict__ g, const float* __restrict__ b,
                          float* __restrict__ out) {
  int i = blockIdx.x * blockDim.x + threadIdx.x;
  int c = i & (C_IN - 1);
  float v = (fp[i] - stats[c]) * stats[C_IN + c] * g[c] + b[c];
  out[i] = fmaxf(v, 0.f);
}

// ---------------------------------------------------------------------- launch
extern "C" void kernel_launch(void* const* d_in, const int* in_sizes, int n_in,
                              void* d_out, int out_size, void* d_ws, size_t ws_size,
                              hipStream_t stream) {
  (void)in_sizes; (void)n_in; (void)out_size; (void)ws_size;
  const float* features = (const float*)d_in[0];
  const int*   coords   = (const int*)d_in[1];
  const float* Wp   = (const float*)d_in[2];
  const float* bp   = (const float*)d_in[3];
  const float* Wpe1 = (const float*)d_in[4];
  const float* bpe1 = (const float*)d_in[5];
  const float* Wpe2 = (const float*)d_in[6];
  const float* bpe2 = (const float*)d_in[7];
  const float* Wq   = (const float*)d_in[8];
  const float* bq   = (const float*)d_in[9];
  const float* Wk   = (const float*)d_in[10];
  const float* bk   = (const float*)d_in[11];
  const float* Wv   = (const float*)d_in[12];
  const float* bv   = (const float*)d_in[13];
  const float* Wo   = (const float*)d_in[14];
  const float* bo   = (const float*)d_in[15];
  const float* W1   = (const float*)d_in[16];
  const float* b1   = (const float*)d_in[17];
  const float* W2   = (const float*)d_in[18];
  const float* b2   = (const float*)d_in[19];
  const float* lng  = (const float*)d_in[20];
  const float* lnb  = (const float*)d_in[21];
  const float* Wf1  = (const float*)d_in[22];
  const float* bf1  = (const float*)d_in[23];
  const float* Wf2  = (const float*)d_in[24];
  const float* bf2  = (const float*)d_in[25];
  const float* bng  = (const float*)d_in[26];
  const float* bnb  = (const float*)d_in[27];
  float* out = (float*)d_out;

  char* p = (char*)d_ws;
  auto take = [&](size_t bytes) -> char* {
    char* q = p; p += (bytes + 255) & ~(size_t)255; return q;
  };
  h16*   srcH   = (h16*)take((size_t)N_PTS * DMODEL * 2);
  int*   cpack  = (int*)take((size_t)N_PTS * 4);
  int*   nidx   = (int*)take((size_t)N_PTS * MNEIGH * 4);
  float* fusedp = (float*)take((size_t)N_PTS * C_IN * 4);
  float* stats  = (float*)take(2 * C_IN * 4);
  h16* WqH  = (h16*)take(128 * 128 * 2);
  h16* WkH  = (h16*)take(128 * 128 * 2);
  h16* WvH  = (h16*)take(128 * 128 * 2);
  h16* WoH  = (h16*)take(128 * 128 * 2);
  h16* W1H  = (h16*)take(256 * 128 * 2);
  h16* W2H  = (h16*)take(128 * 256 * 2);
  h16* Wf1H = (h16*)take(32 * 128 * 2);
  h16* Wf2H = (h16*)take(32 * 64 * 2);

  auto cv = [&](const float* s, h16* d, int n) {
    k_f2h<<<(n + 255) / 256, 256, 0, stream>>>(s, d, n);
  };
  cv(Wq, WqH, 128 * 128);
  cv(Wk, WkH, 128 * 128);
  cv(Wv, WvH, 128 * 128);
  cv(Wo, WoH, 128 * 128);
  cv(W1, W1H, 256 * 128);
  cv(W2, W2H, 128 * 256);
  cv(Wf1, Wf1H, 32 * 128);
  cv(Wf2, Wf2H, 32 * 64);

  k_embed<<<N_PTS, 128, 0, stream>>>(features, coords, Wp, bp, Wpe1, bpe1,
                                     Wpe2, bpe2, srcH, cpack);
  k_neigh<<<N_PTS / 8, 256, 0, stream>>>(cpack, nidx, N_PTS);

  // dynamic LDS:
  //   staged Wk+Wv (64KB) + sK+sV (128KB) + f16 intermediates + f32 residual
  const size_t lds_bytes =
      (2 * 128 * DMODEL +                 // sWk, sWv
       2 * 256 * DMODEL +                 // sK, sV
       16 * DMODEL * 4 +                  // sQ, sAt, sO, sTg
       16 * DFFN + 16 * 64) * sizeof(h16) // sT1, sCat
      + 16 * DMODEL * sizeof(float);      // sO32
  k_attn_block<<<N_PTS / 16, 256, lds_bytes, stream>>>(
      features, srcH, nidx, WqH, bq, WkH, bk, WvH, bv, WoH, bo,
      W1H, b1, W2H, b2, lng, lnb, Wf1H, bf1, Wf2H, bf2, fusedp);

  k_bnstats<<<C_IN, 256, 0, stream>>>(fusedp, stats, N_PTS);
  k_bnapply<<<(N_PTS * C_IN) / 256, 256, 0, stream>>>(fusedp, stats, bng, bnb, out);
}
